// GraphGridCrossAttnModel_34815004901547
// MI455X (gfx1250) — matrix operand order repaired and checked by reading, not compiled, not run on hardware
//
#include <hip/hip_runtime.h>
#include <cstdint>

// ---------------- problem constants ----------------
#define B_   4
#define T_   24
#define N_   2048
#define DG_  64
#define E_   128
#define NH_  4
#define HD_  32
#define SG_  (T_ * N_)     // 49152 graph seq len
#define NP_  64            // patches per frame
#define SR_  (T_ * NP_)    // 1536 grid seq len
#define QROWS (B_ * T_)    // 96
#define NPAIR (NH_ * T_)   // 96 (head,query) pairs per batch
#define PARTW 34           // m, l, o[32]
#define CH_   64           // key rows per flash chunk (keeps LDS ~80KB -> 3-4 WGs/WGP)

typedef __attribute__((ext_vector_type(16))) _Float16 v16h;
typedef __attribute__((ext_vector_type(8)))  float    v8f;
typedef __attribute__((ext_vector_type(4)))  int      v4i_;
typedef _Float16 half_t;

// ---------------- CDNA5 async global->LDS staging (guarded; falls back cleanly) ----
#if __has_builtin(__builtin_amdgcn_global_load_async_to_lds_b128) && \
    __has_builtin(__builtin_amdgcn_s_wait_asynccnt)
#define USE_ASYNC_LDS 1
typedef __attribute__((address_space(1))) v4i_* g4_ptr;   // global v4i*
typedef __attribute__((address_space(3))) v4i_* l4_ptr;   // LDS v4i*
__device__ __forceinline__ void async_copy_b128(const void* g, void* l) {
  // AS-qualified args match either an AS-typed or a generic v4i* builtin signature.
  __builtin_amdgcn_global_load_async_to_lds_b128((g4_ptr)g, (l4_ptr)l, 0, 0);
}
__device__ __forceinline__ void async_wait() { __builtin_amdgcn_s_wait_asynccnt(0); }
#else
#define USE_ASYNC_LDS 0
__device__ __forceinline__ void async_wait() {}
#endif

// ---------------- WMMA fragment helpers (CDNA5 16x16x32 f16 layouts) ----------------
// A (16x32 MxK) from row-major [m][k] tile:
//   lane<16: row=lane, slots0-15 = K 0..15
//   lane>=16: row=lane-16, slots0-15 = K 16..31
__device__ __forceinline__ v16h frag_A(const half_t* p0, int stride) {
  int lane = threadIdx.x & 31;
  const half_t* p = p0 + (lane & 15) * stride + ((lane >> 4) << 4);
  v16h a;
#pragma unroll
  for (int i = 0; i < 8; ++i) a[i] = p[i];
#pragma unroll
  for (int i = 0; i < 8; ++i) a[i + 8] = p[8 + i];
  return a;
}
// B (32x16 KxN) from B^T row-major [n][k] tile (LDS or L2-resident global):
//   lane<16: col n=lane, slots = K 0..15 ; lane>=16: col n=lane-16, slots = K 16..31
__device__ __forceinline__ v16h frag_B(const half_t* p0, int stride) {
  int lane = threadIdx.x & 31;
  const half_t* p = p0 + (lane & 15) * stride + ((lane >> 4) << 4);
  v16h b;
#pragma unroll
  for (int i = 0; i < 16; ++i) b[i] = p[i];
  return b;
}
__device__ __forceinline__ v8f wmma32(v16h a, v16h b, v8f c) {
  return __builtin_amdgcn_wmma_f32_16x16x32_f16(false, a, false, b, (short)0, c, false, false);
}

// Out(64x128 f16 LDS) = act(A(64xK f16 LDS) @ Wt_g^T + bias); Wt_g read straight
// from global (weights are tiny and L2/WGP$ resident). 8 waves: wave&3 -> m-tile,
// wave>>2 -> which half of the 8 n-tiles.
__device__ __forceinline__ void gemm_64x128_gB(const half_t* A, int astr, int K,
                                               const half_t* Wt_g, const float* bias,
                                               half_t* Out, int ostr, bool relu) {
  int wave = threadIdx.x >> 5, lane = threadIdx.x & 31;
  const half_t* Abase = A + (wave & 3) * 16 * astr;
  int nt0 = (wave >> 2) * 4;
#pragma unroll 1
  for (int j = 0; j < 4; ++j) {
    int nt = nt0 + j;
    const half_t* Wb = Wt_g + (size_t)nt * 16 * K;
    v8f c = {};
    for (int kk = 0; kk < K; kk += 32)
      c = wmma32(frag_A(Abase + kk, astr), frag_B(Wb + kk, K), c);
    int n = nt * 16 + (lane & 15);
    float bs = bias[n];
    int rbase = (wave & 3) * 16 + ((lane >> 4) << 3);
#pragma unroll
    for (int i = 0; i < 8; ++i) {
      float x = c[i] + bs;
      if (relu) x = fmaxf(x, 0.0f);
      Out[(rbase + i) * ostr + n] = (half_t)x;
    }
  }
}

// ---------------- tiny prep kernels ----------------
__global__ void te_hidden_kernel(const float* __restrict__ tvals, const float* __restrict__ w1,
                                 const float* __restrict__ b1, float* __restrict__ out) {
  int idx = blockIdx.x * blockDim.x + threadIdx.x;
  if (idx >= QROWS * E_) return;
  int r = idx / E_, k = idx - r * E_;
  out[idx] = fmaxf(tvals[r] * w1[k] + b1[k], 0.0f);
}

__global__ void gemm_small_kernel(const float* __restrict__ A, const float* __restrict__ Wm,
                                  const float* __restrict__ bias, float* __restrict__ out, int M) {
  int idx = blockIdx.x * blockDim.x + threadIdx.x;
  if (idx >= M * E_) return;
  int r = idx / E_, n = idx - r * E_;
  float acc = bias[n];
  const float* a = A + r * E_;
  for (int k = 0; k < E_; ++k) acc = fmaf(a[k], Wm[k * E_ + n], acc);
  out[idx] = acc;
}

// Wt_out[n][h] = (W2@Wo)[h][n] (f16), b_out = b2@Wo + badd
__global__ void fold_kernel(const float* __restrict__ W2, const float* __restrict__ Wo,
                            const float* __restrict__ b2, const float* __restrict__ badd,
                            half_t* __restrict__ Wt_out, float* __restrict__ b_out) {
  int idx = blockIdx.x * blockDim.x + threadIdx.x;
  if (idx >= E_ * E_) return;
  int n = idx / E_, h = idx - n * E_;
  float acc = 0.0f;
  for (int e = 0; e < E_; ++e) acc = fmaf(W2[h * E_ + e], Wo[e * E_ + n], acc);
  Wt_out[n * E_ + h] = (half_t)acc;
  if (h == 0) {
    float bb = badd[n];
    for (int e = 0; e < E_; ++e) bb = fmaf(b2[e], Wo[e * E_ + n], bb);
    b_out[n] = bb;
  }
}

__global__ void transpose_to_f16_kernel(const float* __restrict__ W, half_t* __restrict__ out, int K) {
  int idx = blockIdx.x * blockDim.x + threadIdx.x;
  if (idx >= K * E_) return;
  int n = idx / K, k = idx - n * K;
  out[idx] = (half_t)W[k * E_ + n];
}

__global__ void cast_f16_kernel(const float* __restrict__ in, half_t* __restrict__ out, int n) {
  int idx = blockIdx.x * blockDim.x + threadIdx.x;
  if (idx < n) out[idx] = (half_t)in[idx];
}

__global__ void transpose_up_kernel(const float* __restrict__ up_w, half_t* __restrict__ out) {
  int idx = blockIdx.x * blockDim.x + threadIdx.x;
  if (idx >= 8192 * E_) return;
  int n = idx / E_, c = idx - n * E_;
  out[idx] = (half_t)up_w[c * 8192 + n];
}

// ---------------- fused MLP->K/V->attention flash kernel ----------------
// one workgroup = one 64-row key chunk; folded weights give
// k = relu(src@W1k+b1k)@Wk^T + bk, v = relu(src@W1v+b1v)@Wv^T + bv, all in LDS.
__global__ __launch_bounds__(256) void flash_kernel(
    const float* __restrict__ src, int S, int DIN,
    const half_t* __restrict__ w1k_t, const float* __restrict__ b1k,
    const half_t* __restrict__ w1v_t, const float* __restrict__ b1v,
    const half_t* __restrict__ wk_t, const float* __restrict__ bk,
    const half_t* __restrict__ wv_t, const float* __restrict__ bv,
    const float* __restrict__ qbuf, float* __restrict__ partials, int nchunk) {
  extern __shared__ char smem[];
  const int astr = DIN + 8;
  half_t* sA = (half_t*)smem;             // CH_ x astr (src tile f16)
  half_t* sH = sA + CH_ * astr;           // CH_ x 136 hidden
  half_t* sK = sH + CH_ * 136;            // keys (col = head*32 + d)
  half_t* sV = sK + CH_ * 136;            // values
  float*  sQ = (float*)(sV + CH_ * 136);  // 24 x 128 f32 queries

  const int b = blockIdx.y, chunk = blockIdx.x, tid = threadIdx.x;
  const int row0 = chunk * CH_;

  for (int i = tid; i < T_ * E_; i += 256) sQ[i] = qbuf[b * T_ * E_ + i];
  const float* srow = src + ((size_t)b * S + row0) * DIN;
  for (int i = tid; i < CH_ * DIN; i += 256) {
    int r = i / DIN, c = i - r * DIN;
    sA[r * astr + c] = (half_t)srow[(size_t)r * DIN + c];
  }
  __syncthreads();

  gemm_64x128_gB(sA, astr, DIN, w1k_t, b1k, sH, 136, true);  __syncthreads();
  gemm_64x128_gB(sH, 136, 128, wk_t, bk, sK, 136, false);    __syncthreads();
  gemm_64x128_gB(sA, astr, DIN, w1v_t, b1v, sH, 136, true);  __syncthreads();
  gemm_64x128_gB(sH, 136, 128, wv_t, bv, sV, 136, false);    __syncthreads();

  if (tid < NPAIR) {  // one thread per (head, query): streaming softmax over CH_ keys
    int h = tid / T_, qi = tid - h * T_;
    float qv[HD_];
#pragma unroll
    for (int d = 0; d < HD_; ++d) qv[d] = sQ[qi * E_ + h * HD_ + d];
    const float rscale = 0.17677669529663687f;  // 1/sqrt(32)
    float m = -3.0e38f, l = 0.0f, o[HD_];
#pragma unroll
    for (int d = 0; d < HD_; ++d) o[d] = 0.0f;
    for (int r = 0; r < CH_; ++r) {
      const half_t* kr = sK + r * 136 + h * HD_;
      float s = 0.0f;
#pragma unroll
      for (int d = 0; d < HD_; ++d) s += qv[d] * (float)kr[d];
      s *= rscale;
      float mn = fmaxf(m, s);
      float corr = __expf(m - mn);
      float pr = __expf(s - mn);
      l = l * corr + pr;
      const half_t* vr = sV + r * 136 + h * HD_;
#pragma unroll
      for (int d = 0; d < HD_; ++d) o[d] = o[d] * corr + pr * (float)vr[d];
      m = mn;
    }
    float* pp = partials + (((size_t)b * nchunk + chunk) * NPAIR + tid) * PARTW;
    pp[0] = m; pp[1] = l;
#pragma unroll
    for (int d = 0; d < HD_; ++d) pp[2 + d] = o[d];
  }
}

__global__ void merge_kernel(const float* __restrict__ partials, int nchunk,
                             float* __restrict__ o_att) {
  int idx = blockIdx.x * blockDim.x + threadIdx.x;
  if (idx >= B_ * NPAIR) return;
  int b = idx / NPAIR, pair = idx - b * NPAIR;
  float m = -3.0e38f, l = 0.0f, o[HD_];
#pragma unroll
  for (int d = 0; d < HD_; ++d) o[d] = 0.0f;
  for (int c = 0; c < nchunk; ++c) {
    const float* pp = partials + (((size_t)b * nchunk + c) * NPAIR + pair) * PARTW;
    float mc = pp[0], lc = pp[1];
    float mn = fmaxf(m, mc);
    float ca = __expf(m - mn), cb = __expf(mc - mn);
    l = l * ca + lc * cb;
    for (int d = 0; d < HD_; ++d) o[d] = o[d] * ca + pp[2 + d] * cb;
    m = mn;
  }
  float inv = 1.0f / l;
  int h = pair / T_, qi = pair - h * T_;
  float* out = o_att + ((size_t)b * T_ + qi) * E_ + h * HD_;
  for (int d = 0; d < HD_; ++d) out[d] = o[d] * inv;
}

// ---------------- g_out interp writer: (B,24,128) -> (B,49152,128) ----------------
__global__ void gout_kernel(const float* __restrict__ og, float* __restrict__ out) {
  size_t idx = (size_t)blockIdx.x * blockDim.x + threadIdx.x;
  size_t total = (size_t)B_ * SG_ * E_ / 4;
  if (idx >= total) return;
  size_t f = idx * 4;
  int e = (int)(f & (E_ - 1));
  size_t rj = f >> 7;
  int b = (int)(rj / SG_);
  int j = (int)(rj - (size_t)b * SG_);
  float s = fmaxf((j + 0.5f) * (1.0f / 2048.0f) - 0.5f, 0.0f);
  int i0 = (int)s;
  float lam = s - i0;
  int i1 = min(i0 + 1, T_ - 1);
  const float* r0 = og + ((size_t)b * T_ + i0) * E_ + e;
  const float* r1 = og + ((size_t)b * T_ + i1) * E_ + e;
  float4 v;
  v.x = r0[0] * (1.0f - lam) + r1[0] * lam;
  v.y = r0[1] * (1.0f - lam) + r1[1] * lam;
  v.z = r0[2] * (1.0f - lam) + r1[2] * lam;
  v.w = r0[3] * (1.0f - lam) + r1[3] * lam;
  *(float4*)(out + f) = v;
}

// ---------------- patch embed GEMM: M=6144 rows (64/WG), K=8192 (c,p,q), N=128 -----
__global__ __launch_bounds__(256) void patch_embed_kernel(
    const float* __restrict__ grid, const half_t* __restrict__ pe_h,
    const float* __restrict__ pe_b, float* __restrict__ patches) {
  extern __shared__ char smem[];
  half_t* sA = (half_t*)smem;    // 64 x 72
  half_t* sB = sA + 64 * 72;     // 128 x 72 (pe B^T chunk)
  int tid = threadIdx.x, wave = tid >> 5, lane = tid & 31;
  int mb0 = blockIdx.x * 64;
  // A staging map: 4 threads per row, 16 K (2 p-groups of 8 q) each
  int rloc = tid >> 2;
  int ks = (tid & 3) * 16;
  int m = mb0 + rloc;
  int bt = m >> 6;
  int pp = m & 63;
  int hp = pp >> 3, wp = pp & 7;
  const float* gbase = grid + (size_t)bt * (E_ * 4096) + (hp * 8) * 64 + wp * 8;
  // B staging map: 2 threads per n-row, 32 K each
  int rloc2 = tid >> 1;
  int ks2 = (tid & 1) * 32;

  v8f acc[4] = {};
  for (int c = 0; c < E_; ++c) {
    __syncthreads();
    {  // stage A chunk: vectorized float4 reads of the 201MB grid tensor
      const float* g = gbase + (size_t)c * 4096;
      half_t* dst = sA + rloc * 72 + ks;
#pragma unroll
      for (int pg = 0; pg < 2; ++pg) {
        int p = (ks >> 3) + pg;
        const float4* s4 = (const float4*)(g + p * 64);
        float4 a0 = s4[0], a1 = s4[1];
        half_t* d = dst + pg * 8;
        d[0] = (half_t)a0.x; d[1] = (half_t)a0.y; d[2] = (half_t)a0.z; d[3] = (half_t)a0.w;
        d[4] = (half_t)a1.x; d[5] = (half_t)a1.y; d[6] = (half_t)a1.z; d[7] = (half_t)a1.w;
      }
    }
    {  // stage B chunk: pe_h[o][c*64 + 0..63]
      const half_t* srcp = pe_h + (size_t)rloc2 * 8192 + c * 64 + ks2;
      half_t* dstp = sB + rloc2 * 72 + ks2;
#if USE_ASYNC_LDS
#pragma unroll
      for (int j = 0; j < 4; ++j) async_copy_b128(srcp + j * 8, dstp + j * 8);
#else
#pragma unroll
      for (int j = 0; j < 32; ++j) dstp[j] = srcp[j];
#endif
    }
    async_wait();
    __syncthreads();
    const half_t* Ab = sA + (wave & 3) * 16 * 72;
    int nt0 = (wave >> 2) * 4;
#pragma unroll
    for (int j = 0; j < 4; ++j) {
      const half_t* Bb = sB + (nt0 + j) * 16 * 72;
      acc[j] = wmma32(frag_A(Ab + 0, 72), frag_B(Bb + 0, 72), acc[j]);
      acc[j] = wmma32(frag_A(Ab + 32, 72), frag_B(Bb + 32, 72), acc[j]);
    }
  }
  int rbase = (wave & 3) * 16 + ((lane >> 4) << 3);
  int nt0 = (wave >> 2) * 4;
#pragma unroll 1
  for (int j = 0; j < 4; ++j) {
    int n = (nt0 + j) * 16 + (lane & 15);
    float bs = pe_b[n];
#pragma unroll
    for (int i = 0; i < 8; ++i) {
      int row = mb0 + rbase + i;
      int bt2 = row >> 6;
      int b = bt2 / T_, t = bt2 - b * T_;
      patches[(((size_t)b * SR_) + t * 64 + (row & 63)) * E_ + n] = acc[j][i] + bs;
    }
  }
}

// ---------------- upsample GEMM + fused interp: M=6144 (64/WG), K=128, N=8192 (2048/WG)
__global__ __launch_bounds__(256) void upsample_kernel(
    const float* __restrict__ o_r, const half_t* __restrict__ up_t,
    const float* __restrict__ up_b, float* __restrict__ out) {
  extern __shared__ char smem[];
  half_t* sA = (half_t*)smem;    // 64 x 136 (interp'd r rows, f16)
  half_t* sB = sA + 64 * 136;    // 512 x 136 (up B^T chunk)
  int tid = threadIdx.x, wave = tid >> 5, lane = tid & 31;
  int mb0 = blockIdx.x * 64;
  int nq0 = blockIdx.y * 2048;

  {  // build A rows on the fly from o_r via linear interp (j within batch's 1536)
    int rloc = tid >> 2;
    int cs = (tid & 3) * 32;
    int m = mb0 + rloc;
    int bt = m >> 6;
    int b = bt / T_, t = bt - b * T_;
    int j = t * 64 + (m & 63);
    float s = fmaxf((j + 0.5f) * (1.0f / 64.0f) - 0.5f, 0.0f);
    int i0 = (int)s;
    float lam = s - i0;
    int i1 = min(i0 + 1, T_ - 1);
    const float* r0 = o_r + ((size_t)b * T_ + i0) * E_;
    const float* r1 = o_r + ((size_t)b * T_ + i1) * E_;
    half_t* dst = sA + rloc * 136 + cs;
    for (int c = 0; c < 32; ++c)
      dst[c] = (half_t)(r0[cs + c] * (1.0f - lam) + r1[cs + c] * lam);
  }

  for (int nc = nq0; nc < nq0 + 2048; nc += 512) {
    __syncthreads();
    {  // stage B chunk (512 n-rows x 128 k): async DMA into LDS when available
      int r2 = tid * 2;
      for (int rr = 0; rr < 2; ++rr) {
        const half_t* srcp = up_t + (size_t)(nc + r2 + rr) * 128;
        half_t* dstp = sB + (r2 + rr) * 136;
#if USE_ASYNC_LDS
#pragma unroll
        for (int j = 0; j < 16; ++j) async_copy_b128(srcp + j * 8, dstp + j * 8);
#else
        const uint4* s4 = (const uint4*)srcp;
        uint4* d4 = (uint4*)dstp;
#pragma unroll
        for (int j = 0; j < 16; ++j) d4[j] = s4[j];
#endif
      }
    }
    async_wait();
    __syncthreads();
    const half_t* Ab = sA + (wave & 3) * 16 * 136;
    int rbase0 = mb0 + (wave & 3) * 16 + ((lane >> 4) << 3);
    int t0 = (wave >> 2) * 16;
#pragma unroll 1
    for (int j = 0; j < 16; ++j) {
      int nt = t0 + j;
      const half_t* Bb = sB + nt * 16 * 136;
      v8f c = {};
#pragma unroll
      for (int kk = 0; kk < 128; kk += 32)
        c = wmma32(frag_A(Ab + kk, 136), frag_B(Bb + kk, 136), c);
      int n = nc + nt * 16 + (lane & 15);
      int o = n >> 6, p = (n >> 3) & 7, q = n & 7;
      float bs = up_b[o];
#pragma unroll
      for (int i = 0; i < 8; ++i) {
        int row = rbase0 + i;
        int bt = row >> 6;
        int pq = row & 63;
        int hp = pq >> 3, wp = pq & 7;
        out[((size_t)bt * E_ + o) * 4096 + (hp * 8 + p) * 64 + (wp * 8 + q)] = c[i] + bs;
      }
    }
  }
}

// ---------------- host orchestration ----------------
extern "C" void kernel_launch(void* const* d_in, const int* in_sizes, int n_in,
                              void* d_out, int out_size, void* d_ws, size_t ws_size,
                              hipStream_t stream) {
  (void)in_sizes; (void)n_in; (void)out_size; (void)ws_size;
  const float* gf     = (const float*)d_in[0];
  const float* gridf  = (const float*)d_in[1];
  const float* tvals  = (const float*)d_in[2];  // graph_time_indices drives te
  const float* gek_w1 = (const float*)d_in[4];
  const float* gek_b1 = (const float*)d_in[5];
  const float* gek_w2 = (const float*)d_in[6];
  const float* gek_b2 = (const float*)d_in[7];
  const float* gev_w1 = (const float*)d_in[8];
  const float* gev_b1 = (const float*)d_in[9];
  const float* gev_w2 = (const float*)d_in[10];
  const float* gev_b2 = (const float*)d_in[11];
  const float* rk_w1  = (const float*)d_in[12];
  const float* rk_b1  = (const float*)d_in[13];
  const float* rk_w2  = (const float*)d_in[14];
  const float* rk_b2  = (const float*)d_in[15];
  const float* rv_w1  = (const float*)d_in[16];
  const float* rv_b1  = (const float*)d_in[17];
  const float* rv_w2  = (const float*)d_in[18];
  const float* rv_b2  = (const float*)d_in[19];
  const float* te_w1  = (const float*)d_in[20];
  const float* te_b1  = (const float*)d_in[21];
  const float* te_w2  = (const float*)d_in[22];
  const float* te_b2  = (const float*)d_in[23];
  const float* pe_w   = (const float*)d_in[24];
  const float* pe_b   = (const float*)d_in[25];
  const float* up_w   = (const float*)d_in[26];
  const float* up_b   = (const float*)d_in[27];
  const float* ag_q_w = (const float*)d_in[28];
  const float* ag_q_b = (const float*)d_in[29];
  const float* ag_k_w = (const float*)d_in[30];
  const float* ag_k_b = (const float*)d_in[31];
  const float* ag_v_w = (const float*)d_in[32];
  const float* ag_v_b = (const float*)d_in[33];
  const float* ag_o_w = (const float*)d_in[34];
  const float* ag_o_b = (const float*)d_in[35];
  const float* ar_q_w = (const float*)d_in[36];
  const float* ar_q_b = (const float*)d_in[37];
  const float* ar_k_w = (const float*)d_in[38];
  const float* ar_k_b = (const float*)d_in[39];
  const float* ar_v_w = (const float*)d_in[40];
  const float* ar_v_b = (const float*)d_in[41];
  const float* ar_o_w = (const float*)d_in[42];
  const float* ar_o_b = (const float*)d_in[43];

  uintptr_t cur = (uintptr_t)d_ws;
  auto carve = [&](size_t bytes) -> void* {
    void* p = (void*)cur;
    cur += (bytes + 255) & ~(size_t)255;
    return p;
  };
  const int NCG = SG_ / CH_;  // 768 graph chunks
  const int NCR = SR_ / CH_;  // 24 grid chunks
  float* te_hid   = (float*)carve(QROWS * E_ * 4);
  float* te       = (float*)carve(QROWS * E_ * 4);
  float* q_g      = (float*)carve(QROWS * E_ * 4);
  float* q_r      = (float*)carve(QROWS * E_ * 4);
  float* oatt_g   = (float*)carve(QROWS * E_ * 4);
  float* oatt_r   = (float*)carve(QROWS * E_ * 4);
  float* og       = (float*)carve(QROWS * E_ * 4);
  float* orr      = (float*)carve(QROWS * E_ * 4);
  float* bk_g     = (float*)carve(E_ * 4);
  float* bv_g     = (float*)carve(E_ * 4);
  float* bk_r     = (float*)carve(E_ * 4);
  float* bv_r     = (float*)carve(E_ * 4);
  float* part_g   = (float*)carve((size_t)B_ * NCG * NPAIR * PARTW * 4);
  float* part_r   = (float*)carve((size_t)B_ * NCR * NPAIR * PARTW * 4);
  float* patches  = (float*)carve((size_t)B_ * SR_ * E_ * 4);
  half_t* w1t_g   = (half_t*)carve(E_ * DG_ * 2);
  half_t* w1vt_g  = (half_t*)carve(E_ * DG_ * 2);
  half_t* w1t_r   = (half_t*)carve(E_ * E_ * 2);
  half_t* w1vt_r  = (half_t*)carve(E_ * E_ * 2);
  half_t* wkt_g   = (half_t*)carve(E_ * E_ * 2);
  half_t* wvt_g   = (half_t*)carve(E_ * E_ * 2);
  half_t* wkt_r   = (half_t*)carve(E_ * E_ * 2);
  half_t* wvt_r   = (half_t*)carve(E_ * E_ * 2);
  half_t* pe_h    = (half_t*)carve((size_t)E_ * 8192 * 2);
  half_t* up_t    = (half_t*)carve((size_t)8192 * E_ * 2);

  auto nb = [](size_t n) { return (unsigned)((n + 255) / 256); };

  // time embedding + query projections (tiny)
  te_hidden_kernel<<<nb(QROWS * E_), 256, 0, stream>>>(tvals, te_w1, te_b1, te_hid);
  gemm_small_kernel<<<nb(QROWS * E_), 256, 0, stream>>>(te_hid, te_w2, te_b2, te, QROWS);
  gemm_small_kernel<<<nb(QROWS * E_), 256, 0, stream>>>(te, ag_q_w, ag_q_b, q_g, QROWS);
  gemm_small_kernel<<<nb(QROWS * E_), 256, 0, stream>>>(te, ar_q_w, ar_q_b, q_r, QROWS);

  // fold W2@{Wk,Wv} so K/V come from a single GEMM after the hidden layer
  fold_kernel<<<nb(E_ * E_), 256, 0, stream>>>(gek_w2, ag_k_w, gek_b2, ag_k_b, wkt_g, bk_g);
  fold_kernel<<<nb(E_ * E_), 256, 0, stream>>>(gev_w2, ag_v_w, gev_b2, ag_v_b, wvt_g, bv_g);
  fold_kernel<<<nb(E_ * E_), 256, 0, stream>>>(rk_w2, ar_k_w, rk_b2, ar_k_b, wkt_r, bk_r);
  fold_kernel<<<nb(E_ * E_), 256, 0, stream>>>(rv_w2, ar_v_w, rv_b2, ar_v_b, wvt_r, bv_r);
  transpose_to_f16_kernel<<<nb(DG_ * E_), 256, 0, stream>>>(gek_w1, w1t_g, DG_);
  transpose_to_f16_kernel<<<nb(DG_ * E_), 256, 0, stream>>>(gev_w1, w1vt_g, DG_);
  transpose_to_f16_kernel<<<nb(E_ * E_), 256, 0, stream>>>(rk_w1, w1t_r, E_);
  transpose_to_f16_kernel<<<nb(E_ * E_), 256, 0, stream>>>(rv_w1, w1vt_r, E_);
  cast_f16_kernel<<<nb((size_t)E_ * 8192), 256, 0, stream>>>(pe_w, pe_h, E_ * 8192);
  transpose_up_kernel<<<nb((size_t)8192 * E_), 256, 0, stream>>>(up_w, up_t);

  // graph branch: fused MLP+KV+attention, single pass over graph_features
  {
    size_t smem = (size_t)CH_ * (DG_ + 8) * 2 + 3 * (size_t)CH_ * 136 * 2 + (size_t)T_ * E_ * 4;
    flash_kernel<<<dim3(NCG, B_), 256, smem, stream>>>(
        gf, SG_, DG_, w1t_g, gek_b1, w1vt_g, gev_b1, wkt_g, bk_g, wvt_g, bv_g,
        q_g, part_g, NCG);
  }
  merge_kernel<<<nb(B_ * NPAIR), 256, 0, stream>>>(part_g, NCG, oatt_g);
  gemm_small_kernel<<<nb(QROWS * E_), 256, 0, stream>>>(oatt_g, ag_o_w, ag_o_b, og, QROWS);
  gout_kernel<<<nb((size_t)B_ * SG_ * E_ / 4), 256, 0, stream>>>(og, (float*)d_out);

  // grid branch: patch embed GEMM -> fused attention -> upsample GEMM
  patch_embed_kernel<<<96, 256, (size_t)(64 * 72 + 128 * 72) * 2, stream>>>(gridf, pe_h, pe_b, patches);
  {
    size_t smem = (size_t)CH_ * (E_ + 8) * 2 + 3 * (size_t)CH_ * 136 * 2 + (size_t)T_ * E_ * 4;
    flash_kernel<<<dim3(NCR, B_), 256, smem, stream>>>(
        patches, SR_, E_, w1t_r, rk_b1, w1vt_r, rv_b1, wkt_r, bk_r, wvt_r, bv_r,
        q_r, part_r, NCR);
  }
  merge_kernel<<<nb(B_ * NPAIR), 256, 0, stream>>>(part_r, NCR, oatt_r);
  gemm_small_kernel<<<nb(QROWS * E_), 256, 0, stream>>>(oatt_r, ar_o_w, ar_o_b, orr, QROWS);
  {
    size_t smem = (size_t)(64 * 136 + 512 * 136) * 2;
    float* out_up = (float*)d_out + (size_t)B_ * SG_ * E_;
    upsample_kernel<<<dim3(96, 4), 256, smem, stream>>>(orr, up_t, up_b, out_up);
  }
}
